// PointNet_79517024519023
// MI455X (gfx1250) — matrix operand order repaired and checked by reading, not compiled, not run on hardware
//
#include <hip/hip_runtime.h>
#include <hip/hip_bf16.h>

typedef __attribute__((ext_vector_type(16))) __bf16       v16bf;
typedef __attribute__((ext_vector_type(8)))  float        v8f;
typedef __attribute__((ext_vector_type(8)))  unsigned int v8u;
typedef __attribute__((ext_vector_type(4)))  unsigned int v4u;

#define NGRAPH 256
#define DOUT   1024
#define BN_EPS 1e-5f

#define BLK        256                 // threads per block (8 wave32)
#define PTS_BLOCK  256                 // points per block (32 per wave, 2 M-tiles)
#define NCHUNK     8                   // n-tiles (16 cols each) per W4 LDS chunk -> 128 cols
#define NCHUNKS    8                   // 1024 / 128

#define LDP2       72                  // padded LDS row stride (elements), W2  (144B)
#define LDP3       136                 // padded LDS row stride (elements), W3  (272B)
#define LDP4       136                 // padded LDS row stride (elements), W4 chunk

#define SCRATCH_SH (8 * 2048)          // per-wave h2/h3 scratch (overlapped)
#define W2_SH      (64 * LDP2)         // 4608
#define W3_SH      (64 * LDP3)         // 8704
#define CHUNK_SH   (128 * LDP4)        // 17408 per W4 buffer
#define OFF_W2     SCRATCH_SH
#define OFF_W3     (OFF_W2 + W2_SH)
#define OFF_W4A    (OFF_W3 + W3_SH)
#define OFF_W4B    (OFF_W4A + CHUNK_SH)
#define SMEM_SH    (OFF_W4B + CHUNK_SH)   // 64512 shorts = 126 KB

// ---------- helpers ----------

__device__ __forceinline__ unsigned short bfbits(float x) {
  unsigned u = __float_as_uint(x);
  u += 0x7FFFu + ((u >> 16) & 1u);
  return (unsigned short)(u >> 16);
}

__device__ __forceinline__ unsigned short bfst(float x) {   // native v_cvt path
  return __builtin_bit_cast(unsigned short, (__bf16)x);
}

// 16-half fragment as two 16B loads: halves 0-7 at p, halves 8-15 at p+gap (elements).
// gap=8 -> contiguous (B-fragment row slice); gap=16 -> A-fragment (K, K+16 blocks).
__device__ __forceinline__ v16bf frag_gap(const unsigned short* p, int gap) {
  v4u a = *(const v4u*)p;
  v4u b = *(const v4u*)(p + gap);
  v8u r;
  r[0] = a[0]; r[1] = a[1]; r[2] = a[2]; r[3] = a[3];
  r[4] = b[0]; r[5] = b[1]; r[6] = b[2]; r[7] = b[3];
  return __builtin_bit_cast(v16bf, r);
}

// one async 16B global->LDS copy (GVS addressing: SGPR base + 32-bit VGPR offset)
__device__ __forceinline__ void async_b128(const void* gbase, unsigned goff, unsigned loff) {
  asm volatile("global_load_async_to_lds_b128 %0, %1, %2"
               :: "v"(loff), "v"(goff), "s"(gbase) : "memory");
}

// float atomic max via integer atomics (valid with -inf init)
__device__ __forceinline__ void atomicMaxF(float* addr, float val) {
  if (val >= 0.0f) atomicMax((int*)addr, __float_as_int(val));
  else             atomicMin((unsigned int*)addr, __float_as_uint(val));
}

// ---------- prep: fold BN into weights, bf16-ify, init output to -inf ----------

__global__ void prep_kernel(const float* __restrict__ W1, const float* __restrict__ b1,
                            const float* __restrict__ g1, const float* __restrict__ be1,
                            const float* __restrict__ m1, const float* __restrict__ v1,
                            const float* __restrict__ W2, const float* __restrict__ b2,
                            const float* __restrict__ W3, const float* __restrict__ b3,
                            const float* __restrict__ g3, const float* __restrict__ be3,
                            const float* __restrict__ m3, const float* __restrict__ v3,
                            const float* __restrict__ W4, const float* __restrict__ b4,
                            float* __restrict__ W1f, float* __restrict__ b1f,
                            float* __restrict__ b2f, float* __restrict__ b3f,
                            float* __restrict__ b4f,
                            unsigned short* __restrict__ W2bf,
                            unsigned short* __restrict__ W3bf,
                            unsigned short* __restrict__ W4bf,
                            float* __restrict__ out) {
  const int i = blockIdx.x * blockDim.x + threadIdx.x;
  if (i < NGRAPH * DOUT) out[i] = -__builtin_inff();
  if (i < 64) {
    float s = rsqrtf(v1[i] + BN_EPS) * g1[i];
    b1f[i]       = (b1[i] - m1[i]) * s + be1[i];
    W1f[i]       = W1[i]       * s;
    W1f[64 + i]  = W1[64 + i]  * s;
    W1f[128 + i] = W1[128 + i] * s;
    b2f[i]       = b2[i];
  }
  if (i < 128) {
    float s = rsqrtf(v3[i] + BN_EPS) * g3[i];
    b3f[i] = (b3[i] - m3[i]) * s + be3[i];
  }
  if (i < 1024) b4f[i] = b4[i];
  if (i < 64 * 64) W2bf[i] = bfbits(W2[i]);
  if (i < 64 * 128) {
    int n = i & 127;
    float s = rsqrtf(v3[n] + BN_EPS) * g3[n];
    W3bf[i] = bfbits(W3[i] * s);
  }
  if (i < 128 * 1024) W4bf[i] = bfbits(W4[i]);
}

// ---------- fused pipeline ----------

__global__ void __launch_bounds__(256)
pointnet_fused(const float* __restrict__ x, const int* __restrict__ batch, int N,
               const float* __restrict__ W1f, const float* __restrict__ b1f,
               const unsigned short* __restrict__ W2bf, const float* __restrict__ b2f,
               const unsigned short* __restrict__ W3bf, const float* __restrict__ b3f,
               const unsigned short* __restrict__ W4bf, const float* __restrict__ b4f,
               float* __restrict__ out) {
  __shared__ unsigned short smem[SMEM_SH];
  const int lane = threadIdx.x & 31;
  const int wave = threadIdx.x >> 5;
  const int row  = lane & 15;
  const int half = lane >> 4;
  unsigned short* hscr  = smem + wave * 2048;     // h2 [16][64] then h3 [16][128]
  unsigned short* w2lds = smem + OFF_W2;          // [64][LDP2]
  unsigned short* w3lds = smem + OFF_W3;          // [64][LDP3]
  unsigned short* w4buf0 = smem + OFF_W4A;        // [128][LDP4]
  unsigned short* w4buf1 = smem + OFF_W4B;

  const int tid = (int)threadIdx.x;
  const int waveBase = blockIdx.x * PTS_BLOCK + wave * 32;

  // W4 chunk copy: 2048 16B-units per 128x128 chunk, 8 per thread
  auto issueChunk = [&](int c, unsigned short* dst) {
#pragma unroll
    for (int j = 0; j < 8; ++j) {
      int u  = tid + BLK * j;
      int r  = u >> 4, cu = u & 15;
      async_b128(W4bf, (unsigned)(r * 2048 + c * 256 + cu * 16),
                 (unsigned)(uintptr_t)(dst + r * LDP4 + cu * 8));
    }
  };

  // ---- kick off all block-shared copies; stage 1 runs underneath them ----
  {
#pragma unroll
    for (int j = 0; j < 2; ++j) {                 // W2: 512 units
      int u = tid + BLK * j;
      int r = u >> 3, cu = u & 7;
      async_b128(W2bf, (unsigned)(r * 128 + cu * 16),
                 (unsigned)(uintptr_t)(w2lds + r * LDP2 + cu * 8));
    }
#pragma unroll
    for (int j = 0; j < 4; ++j) {                 // W3: 1024 units
      int u = tid + BLK * j;
      int r = u >> 4, cu = u & 15;
      async_b128(W3bf, (unsigned)(r * 256 + cu * 16),
                 (unsigned)(uintptr_t)(w3lds + r * LDP3 + cu * 8));
    }
    issueChunk(0, w4buf0);
    issueChunk(1, w4buf1);                        // 22 async ops in flight per wave
  }

  const int g0  = batch[min(waveBase, N - 1)];
  const int g31 = batch[min(waveBase + 31, N - 1)];
  const bool uni = (g0 == g31);                   // wave-uniform (sorted batch, ~2k pts/graph)

  // ================= stage 1 for both M-tiles (overlaps async copies) =================
  v16bf a1m[2][2];
#pragma unroll
  for (int mt = 0; mt < 2; ++mt) {
    const int p = min(waveBase + mt * 16 + row, N - 1);   // tail duplicates last pt (max-safe)
    const float x0 = x[3 * p + 0], x1 = x[3 * p + 1], x2 = x[3 * p + 2];
#pragma unroll
    for (int kt = 0; kt < 2; ++kt) {
#pragma unroll
      for (int h = 0; h < 16; ++h) {
        int kk = kt * 32 + ((h & 8) ? 16 : 0) + half * 8 + (h & 7);
        float val = fmaf(x0, W1f[kk], fmaf(x1, W1f[64 + kk], fmaf(x2, W1f[128 + kk], b1f[kk])));
        a1m[mt][kt][h] = (__bf16)fmaxf(val, 0.0f);
      }
    }
  }

  asm volatile("s_wait_asynccnt 0x10" ::: "memory");      // W2+W3 landed; W4 c0,c1 in flight
  __syncthreads();

  // ================= stages 2-3 (W2/W3 from LDS) =================
  v16bf a3[2][4];                                         // h3 A-fragments, K=128
#pragma unroll
  for (int mt = 0; mt < 2; ++mt) {
    // stage 2: h2 = h1 @ W2 + b2 -> LDS bf16 [16][64]
#pragma unroll
    for (int nt = 0; nt < 4; ++nt) {
      v8f acc = {0.f, 0.f, 0.f, 0.f, 0.f, 0.f, 0.f, 0.f};
#pragma unroll
      for (int kt = 0; kt < 2; ++kt) {
        v16bf b;
        for (int s = 0; s < 16; ++s)
          b[s] = __builtin_bit_cast(__bf16, w2lds[(kt * 32 + ((s & 8) ? 16 : 0) + half * 8 + (s & 7)) * LDP2 + nt * 16 + row]);
        acc = __builtin_amdgcn_wmma_f32_16x16x32_bf16(false, a1m[mt][kt], false, b,
                                                      (short)0, acc, false, false);
      }
      float bias = b2f[nt * 16 + row];
#pragma unroll
      for (int v = 0; v < 8; ++v)
        hscr[(v + half * 8) * 64 + nt * 16 + row] = bfst(acc[v] + bias);
    }

    v16bf a2[2];                                          // h2 A-fragments (buffer then free)
#pragma unroll
    for (int kt = 0; kt < 2; ++kt)
      a2[kt] = frag_gap(hscr + row * 64 + kt * 32 + half * 8, 16);

    // stage 3: h3 = relu(h2 @ W3' + b3') -> LDS bf16 [16][128]
#pragma unroll
    for (int nt = 0; nt < 8; ++nt) {
      v8f acc = {0.f, 0.f, 0.f, 0.f, 0.f, 0.f, 0.f, 0.f};
#pragma unroll
      for (int kt = 0; kt < 2; ++kt) {
        v16bf b;
        for (int s = 0; s < 16; ++s)
          b[s] = __builtin_bit_cast(__bf16, w3lds[(kt * 32 + ((s & 8) ? 16 : 0) + half * 8 + (s & 7)) * LDP3 + nt * 16 + row]);
        acc = __builtin_amdgcn_wmma_f32_16x16x32_bf16(false, a2[kt], false, b,
                                                      (short)0, acc, false, false);
      }
      float bias = b3f[nt * 16 + row];
#pragma unroll
      for (int v = 0; v < 8; ++v)
        hscr[(v + half * 8) * 128 + nt * 16 + row] = bfst(fmaxf(acc[v] + bias, 0.0f));
    }
#pragma unroll
    for (int kt = 0; kt < 4; ++kt)
      a3[mt][kt] = frag_gap(hscr + row * 128 + kt * 32 + half * 8, 16);
  }

  // ================= stage 4: h3 @ W4 + b4 with fused segment-max =================
  for (int c = 0; c < NCHUNKS; ++c) {
    unsigned short* cur = (c & 1) ? w4buf1 : w4buf0;
    if (c < NCHUNKS - 1) asm volatile("s_wait_asynccnt 0x8" ::: "memory");  // chunk c landed
    else                 asm volatile("s_wait_asynccnt 0x0" ::: "memory");
    __syncthreads();

#pragma unroll
    for (int ntl = 0; ntl < NCHUNK; ++ntl) {
      v8f acc0 = {0.f, 0.f, 0.f, 0.f, 0.f, 0.f, 0.f, 0.f};
      v8f acc1 = {0.f, 0.f, 0.f, 0.f, 0.f, 0.f, 0.f, 0.f};
#pragma unroll
      for (int kt = 0; kt < 4; ++kt) {
        v16bf b;
        for (int s = 0; s < 16; ++s)
          b[s] = __builtin_bit_cast(__bf16, cur[(kt * 32 + ((s & 8) ? 16 : 0) + half * 8 + (s & 7)) * LDP4 + ntl * 16 + row]);
        acc0 = __builtin_amdgcn_wmma_f32_16x16x32_bf16(false, a3[0][kt], false, b,
                                                       (short)0, acc0, false, false);
        acc1 = __builtin_amdgcn_wmma_f32_16x16x32_bf16(false, a3[1][kt], false, b,
                                                       (short)0, acc1, false, false);
      }
      const int n = (c * NCHUNK + ntl) * 16 + row;
      const float bias = b4f[n];
      if (uni) {
        float m = fmaxf(acc0[0], acc1[0]);
#pragma unroll
        for (int v = 1; v < 8; ++v) m = fmaxf(m, fmaxf(acc0[v], acc1[v]));
        m += bias;
        m = fmaxf(m, __shfl_xor(m, 16, 32));      // combine row-halves across lane pairs
        if (lane < 16) atomicMaxF(out + g0 * DOUT + n, m);
      } else {
#pragma unroll
        for (int v = 0; v < 8; ++v) {
          int p0 = min(waveBase + v + half * 8, N - 1);
          int p1 = min(waveBase + 16 + v + half * 8, N - 1);
          atomicMaxF(out + batch[p0] * DOUT + n, acc0[v] + bias);
          atomicMaxF(out + batch[p1] * DOUT + n, acc1[v] + bias);
        }
      }
    }
    __syncthreads();                               // all waves done reading `cur`
    if (c + 2 < NCHUNKS) issueChunk(c + 2, cur);   // refill the just-freed buffer
  }
}

// ---------- launch ----------

extern "C" void kernel_launch(void* const* d_in, const int* in_sizes, int n_in,
                              void* d_out, int out_size, void* d_ws, size_t ws_size,
                              hipStream_t stream) {
  (void)n_in; (void)out_size; (void)ws_size;
  const float* x   = (const float*)d_in[0];
  const int* batch = (const int*)d_in[1];
  const float* W1  = (const float*)d_in[2];
  const float* b1  = (const float*)d_in[3];
  const float* g1  = (const float*)d_in[4];
  const float* be1 = (const float*)d_in[5];
  const float* m1  = (const float*)d_in[6];
  const float* v1  = (const float*)d_in[7];
  const float* W2  = (const float*)d_in[8];
  const float* b2  = (const float*)d_in[9];
  const float* W3  = (const float*)d_in[10];
  const float* b3  = (const float*)d_in[11];
  const float* g3  = (const float*)d_in[12];
  const float* be3 = (const float*)d_in[13];
  const float* m3  = (const float*)d_in[14];
  const float* v3  = (const float*)d_in[15];
  const float* W4  = (const float*)d_in[16];
  const float* b4  = (const float*)d_in[17];
  const int N = in_sizes[1];

  char* ws = (char*)d_ws;                               // 256B-aligned offsets
  float*          W1f  = (float*)(ws + 0);              // 3*64 f32
  float*          b1f  = (float*)(ws + 1024);           // 64 f32
  float*          b2f  = (float*)(ws + 1280);           // 64 f32
  float*          b3f  = (float*)(ws + 1536);           // 128 f32
  float*          b4f  = (float*)(ws + 2048);           // 1024 f32
  unsigned short* W2bf = (unsigned short*)(ws + 6144);  // 64*64 bf16
  unsigned short* W3bf = (unsigned short*)(ws + 14336); // 64*128 bf16
  unsigned short* W4bf = (unsigned short*)(ws + 30720); // 128*1024 bf16
  float* out = (float*)d_out;

  prep_kernel<<<(NGRAPH * DOUT + 255) / 256, 256, 0, stream>>>(
      W1, b1, g1, be1, m1, v1, W2, b2, W3, b3, g3, be3, m3, v3, W4, b4,
      W1f, b1f, b2f, b3f, b4f, W2bf, W3bf, W4bf, out);

  pointnet_fused<<<(N + PTS_BLOCK - 1) / PTS_BLOCK, BLK, 0, stream>>>(
      x, batch, N, W1f, b1f, W2bf, b2f, W3bf, b3f, W4bf, b4f, out);
}
